// EdgeConv_23648089932273
// MI455X (gfx1250) — compile-verified, matrix operand order checked by reading, not run on hardware
//
#include <hip/hip_runtime.h>

typedef _Float16 v16h __attribute__((ext_vector_type(16)));
typedef _Float16 v8h  __attribute__((ext_vector_type(8)));
typedef float    v8f  __attribute__((ext_vector_type(8)));
typedef float    v4f  __attribute__((ext_vector_type(4)));

#define T_TOK 32
#define DIM   128
#define HID   64
#define BLOCK 256
#define WAVES_PER_BLOCK (BLOCK / 32)
#define SSTR  20   // score-scratch row stride in floats (pad: halves hit disjoint banks)

// Involutive lane<->token permutation: swaps [8..15] <-> [16..23]
#define PERM(x) ((x) < 8 ? (x) : (x) < 16 ? (x) + 8 : (x) < 24 ? (x) - 8 : (x))

__device__ __forceinline__ float bcast_lane(float v, int srcLane) {
    return __int_as_float(__builtin_amdgcn_readlane(__float_as_int(v), srcLane));
}

__global__ __launch_bounds__(BLOCK)
void edgeconv_attn_wmma(const float* __restrict__ feats,
                        const float* __restrict__ W1,
                        const float* __restrict__ b1,
                        const float* __restrict__ W2,
                        const float* __restrict__ b2,
                        float* __restrict__ out,
                        int n_total)
{
    // W1 transposed to f16: w1t[h][k], h-major, stride DIM halfs (16 KB)
    __shared__ _Float16 w1t[HID * DIM];
    // Per-wave score transpose scratch: [t][col], padded rows (20 KB total)
    __shared__ float sscr[WAVES_PER_BLOCK][T_TOK * SSTR];

    const int tid = threadIdx.x;
    for (int i = tid; i < HID * DIM; i += BLOCK) {
        const int k = i & (DIM - 1);
        const int h = i >> 7;
        w1t[h * DIM + k] = (_Float16)W1[k * HID + h];
    }
    __syncthreads();

    const int lane   = tid & 31;
    const int col    = lane & 15;     // local N / M within 16
    const int hiHalf = lane >> 4;     // 0: lanes 0-15, 1: lanes 16-31
    const int kShift = hiHalf << 3;   // K-group offset per ISA fragment striping
    const int wave   = tid >> 5;

    const int wave_gid = blockIdx.x * WAVES_PER_BLOCK + wave;
    const int wave_cnt = gridDim.x * WAVES_PER_BLOCK;

    // Per-lane column constants (invariant over n)
    float b1v[4], w2v[4];
    #pragma unroll
    for (int nT = 0; nT < 4; ++nT) {
        b1v[nT] = b1[nT * 16 + col];
        w2v[nT] = W2[nT * 16 + col];
    }

    // Row this lane will read back from the score scratch (= PERM(lane))
    const int trow = PERM(lane);
    float* const srow = sscr[wave];
    const float* const rrow = srow + trow * SSTR;

    const v8f vzero8 = {0.f, 0.f, 0.f, 0.f, 0.f, 0.f, 0.f, 0.f};

    for (int n = wave_gid; n < n_total; n += wave_cnt) {
        const float* fbase = feats + (size_t)n * (T_TOK * DIM);

        // Prefetch next row's tile (speculative)
        if (n + wave_cnt < n_total) {
            const float* fnext = feats + (size_t)(n + wave_cnt) * (T_TOK * DIM);
            __builtin_prefetch(fnext + lane * DIM, 0, 1);
        }

        // Opaque zero: stops LICM from hoisting all 16 B fragments into VGPRs
        unsigned opq = 0;
        asm volatile("" : "+v"(opq));
        const _Float16* const w1p = w1t + opq;

        // ---- GEMM: h = feats_n (32x128) @ W1 (128x64), f16 WMMA, f32 acc ----
        v8f acc[2][4];
        #pragma unroll
        for (int m = 0; m < 2; ++m)
            #pragma unroll
            for (int nT = 0; nT < 4; ++nT)
                acc[m][nT] = vzero8;

        #pragma unroll
        for (int kk = 0; kk < 4; ++kk) {
            // A fragments (16x32 f16): global f32 loads + cvt
            v16h afrag[2];
            #pragma unroll
            for (int m = 0; m < 2; ++m) {
                const float* p = fbase + (m * 16 + col) * DIM + kk * 32 + kShift;
                const v4f f0 = *(const v4f*)(p);
                const v4f f1 = *(const v4f*)(p + 4);
                const v4f f2 = *(const v4f*)(p + 16);
                const v4f f3 = *(const v4f*)(p + 20);
                v16h a;
                #pragma unroll
                for (int j = 0; j < 4; ++j) {
                    a[j]      = (_Float16)f0[j];
                    a[j + 4]  = (_Float16)f1[j];
                    a[j + 8]  = (_Float16)f2[j];
                    a[j + 12] = (_Float16)f3[j];
                }
                afrag[m] = a;
            }
            // B fragments (32x16 f16) from LDS, two ds_load_b128 each
            #pragma unroll
            for (int nT = 0; nT < 4; ++nT) {
                const _Float16* bp = w1p + (nT * 16 + col) * DIM + kk * 32 + kShift;
                const v8h blo = *(const v8h*)(bp);
                const v8h bhi = *(const v8h*)(bp + 16);
                v16h bfrag;
                #pragma unroll
                for (int j = 0; j < 8; ++j) { bfrag[j] = blo[j]; bfrag[j + 8] = bhi[j]; }
                acc[0][nT] = __builtin_amdgcn_wmma_f32_16x16x32_f16(
                    false, afrag[0], false, bfrag, (short)0, acc[0][nT], false, false);
                acc[1][nT] = __builtin_amdgcn_wmma_f32_16x16x32_f16(
                    false, afrag[1], false, bfrag, (short)0, acc[1][nT], false, false);
            }
        }

        // ---- per-(row,col) score partials -> LDS transpose ----
        // C/D layout: vgpr r, lane L -> row M = r + 8*(L>=16), col N = L%16
        // token t = 16*m + r + 8*hiHalf ; this lane contributes column `col`
        #pragma unroll
        for (int m = 0; m < 2; ++m) {
            #pragma unroll
            for (int r = 0; r < 8; ++r) {
                float s = 0.f;
                #pragma unroll
                for (int nT = 0; nT < 4; ++nT) {
                    float v = acc[m][nT][r] + b1v[nT];
                    v = v > 0.f ? v : 0.f;
                    s += v * w2v[nT];
                }
                srow[(16 * m + r + 8 * hiHalf) * SSTR + col] = s;
            }
        }

        // Read back one full token row per lane (same-wave DS ops are in-order)
        const v4f s0 = *(const v4f*)(rrow);
        const v4f s1 = *(const v4f*)(rrow + 4);
        const v4f s2 = *(const v4f*)(rrow + 8);
        const v4f s3 = *(const v4f*)(rrow + 12);
        const v4f sv = (s0 + s1) + (s2 + s3);
        const float score = (sv[0] + sv[1]) + (sv[2] + sv[3]); // + b2: softmax-invariant

        // ---- softmax across 32 lanes (permutation-invariant reductions) ----
        float mx = score;
        mx = fmaxf(mx, __shfl_xor(mx, 1, 32));
        mx = fmaxf(mx, __shfl_xor(mx, 2, 32));
        mx = fmaxf(mx, __shfl_xor(mx, 4, 32));
        mx = fmaxf(mx, __shfl_xor(mx, 8, 32));
        mx = fmaxf(mx, __shfl_xor(mx, 16, 32));
        const float e = __expf(score - mx);
        float ssum = e;
        ssum += __shfl_xor(ssum, 1, 32);
        ssum += __shfl_xor(ssum, 2, 32);
        ssum += __shfl_xor(ssum, 4, 32);
        ssum += __shfl_xor(ssum, 8, 32);
        ssum += __shfl_xor(ssum, 16, 32);
        const float attn_lane = e * (1.f / ssum);  // attn for token PERM(lane)

        // ---- out[n, 4*lane .. 4*lane+3] = sum_t attn[t] * feats[n,t,:] ----
        // attn[t] lives on lane PERM(t); broadcast to SGPR via v_readlane
        const float* fcol = fbase + lane * 4;
        v4f o = {0.f, 0.f, 0.f, 0.f};
        #pragma unroll
        for (int t = 0; t < T_TOK; ++t) {
            const float at = bcast_lane(attn_lane, PERM(t));
            const v4f f = *(const v4f*)(fcol + t * DIM);
            o += at * f;
        }
        *(v4f*)(out + (size_t)n * DIM + lane * 4) = o;
    }
}

extern "C" void kernel_launch(void* const* d_in, const int* in_sizes, int n_in,
                              void* d_out, int out_size, void* d_ws, size_t ws_size,
                              hipStream_t stream) {
    const float* feats = (const float*)d_in[0];
    const float* W1    = (const float*)d_in[1];
    const float* b1    = (const float*)d_in[2];
    const float* W2    = (const float*)d_in[3];
    const float* b2    = (const float*)d_in[4];
    (void)b2;
    float* out = (float*)d_out;

    const int n_total = in_sizes[0] / (T_TOK * DIM);

    const int rows_per_wave = 4;
    int blocks = (n_total + WAVES_PER_BLOCK * rows_per_wave - 1) /
                 (WAVES_PER_BLOCK * rows_per_wave);
    if (blocks < 1) blocks = 1;

    edgeconv_attn_wmma<<<blocks, BLOCK, 0, stream>>>(feats, W1, b1, W2, b2, out, n_total);
}
